// GRecRTS_6236292514591
// MI455X (gfx1250) — matrix operand order repaired and checked by reading, not compiled, not run on hardware
//
#include <hip/hip_runtime.h>
#include <cstdint>
#include <cstddef>

// ---------------- problem constants ----------------
#define N_TOK   16384
#define DIM     768
#define NEXP    8
#define TOPK    2
#define CAP     5120
#define DHALF   384

typedef __attribute__((ext_vector_type(16))) __bf16 bf16x16;
typedef __attribute__((ext_vector_type(8)))  float  floatx8;

union FragAB { bf16x16 v; uint4 q[2]; };

// =====================================================================
// Generic bf16 WMMA GEMM:  Out = act(A @ B + bias)
//  MODE 0: relu  -> bf16 out          (expert layer 1)
//  MODE 1: none  -> bf16 out          (expert layer 2)
//  MODE 2: leaky(0.2) -> bf16 out     (task head layer 0)
//  MODE 3: row-masked -> f32 out, col bounds (task head layer 1)
//  GATHER:   A rows gathered from fp32 matrix via garray (convert to bf16)
//  BALIGNED: Nsize%8==0 && ldb%8==0 -> vectorized B staging (no scalar path)
// Block tile 128x128, BK=32, 256 threads = 8 waves (4Mx2N), each wave a
// 32x64 subtile = 2x4 v_wmma_f32_16x16x32_bf16 per K-step.
// A operand is staged global->LDS with async-to-LDS DMA (ASYNCcnt),
// B is staged through VGPRs because it must be transposed for the
// B-fragment layout (sB[n][k] contiguous in k).
// =====================================================================
template<int MODE, bool GATHER, bool BALIGNED>
__global__ __launch_bounds__(256)
void wmma_gemm_kernel(const __bf16* __restrict__ Abf,
                      const float*  __restrict__ Af32,
                      const int*    __restrict__ garray,
                      const __bf16* __restrict__ Bmat,
                      const float*  __restrict__ bias,
                      void*         __restrict__ OutV,
                      const int*    __restrict__ valid,
                      int Nsize, int Ksize, int lda, int ldb, int ldo,
                      long long strideA, long long strideB,
                      long long strideBias, long long strideO,
                      const int* __restrict__ rowCount, int Mcap)
{
    constexpr int BM = 128, BN = 128, BK = 32, LSTR = 40; // pad vs bank conflicts
    __shared__ __bf16 sA[BM * LSTR];   // sA[m][k]
    __shared__ __bf16 sB[BN * LSTR];   // transposed: sB[n][k]

    const int tid   = threadIdx.x;
    const int bz    = blockIdx.z;
    const int tileM = blockIdx.y * BM;
    const int tileN = blockIdx.x * BN;

    if (rowCount) {                    // expert GEMMs: skip tiles past routed rows
        int cnt = rowCount[bz]; if (cnt > Mcap) cnt = Mcap;
        if (tileM >= cnt) return;
    }

    const __bf16* Ab = GATHER ? (const __bf16*)nullptr : (Abf + (size_t)bz * strideA);
    const __bf16* Bb = Bmat + (size_t)bz * strideB;
    const float*  bb = bias + (size_t)bz * strideBias;

    const int  wave   = tid >> 5;
    const int  lane   = tid & 31;
    const int  wm     = (wave & 3) * 32;   // 4 waves along M
    const int  wn     = (wave >> 2) * 64;  // 2 waves along N
    const int  lmod   = lane & 15;
    const bool hiHalf = lane >= 16;

    floatx8 zeroA = {0.f,0.f,0.f,0.f,0.f,0.f,0.f,0.f};
    floatx8 acc[2][4];
    #pragma unroll
    for (int i = 0; i < 2; ++i)
        #pragma unroll
        for (int j = 0; j < 4; ++j) acc[i][j] = zeroA;

    for (int k0 = 0; k0 < Ksize; k0 += BK) {
        __syncthreads();
        // ---- stage A tile (128x32 bf16) ----
        if (!GATHER) {
            // Async DMA global->LDS (no VGPR round-trip); tracked on ASYNCcnt.
            #pragma unroll
            for (int c = 0; c < 2; ++c) {
                int ch = tid * 2 + c;                 // 512 chunks of 8 bf16
                int r  = ch >> 2;
                int kk = (ch & 3) * 8;
                const __bf16* g = Ab + (size_t)(tileM + r) * lda + k0 + kk;
                unsigned ldsAddr = (unsigned)(uintptr_t)(&sA[r * LSTR + kk]);
                asm volatile("global_load_async_to_lds_b128 %0, %1, off"
                             :: "v"(ldsAddr), "v"(g) : "memory");
            }
        } else {
            #pragma unroll
            for (int c = 0; c < 4; ++c) {
                int ch  = c * 256 + tid;              // 1024 chunks of 4 floats
                int r   = ch >> 3;
                int kk  = (ch & 7) * 4;
                int grow = garray[tileM + r];
                float4 v = *(const float4*)(Af32 + (size_t)grow * lda + k0 + kk);
                __bf16 o[4] = {(__bf16)v.x, (__bf16)v.y, (__bf16)v.z, (__bf16)v.w};
                *(uint2*)(&sA[r * LSTR + kk]) = *(const uint2*)o;
            }
        }
        // ---- stage B tile transposed (128 cols x 32 k) ----
        #pragma unroll
        for (int c = 0; c < 2; ++c) {
            int ch  = c * 256 + tid;                  // 512 chunks of 8 bf16
            int kr  = ch >> 4;
            int n0  = (ch & 15) * 8;
            int gn0 = tileN + n0;
            __bf16 tb[8];
            if (BALIGNED) {
                if (gn0 + 8 <= Nsize) {
                    uint4 v = *(const uint4*)(Bb + (size_t)(k0 + kr) * ldb + gn0);
                    *(uint4*)tb = v;
                } else {
                    #pragma unroll
                    for (int j = 0; j < 8; ++j) tb[j] = (__bf16)0.0f;
                }
            } else {
                #pragma unroll
                for (int j = 0; j < 8; ++j)
                    tb[j] = (gn0 + j < Nsize) ? Bb[(size_t)(k0 + kr) * ldb + gn0 + j]
                                              : (__bf16)0.0f;
            }
            #pragma unroll
            for (int j = 0; j < 8; ++j) sB[(n0 + j) * LSTR + kr] = tb[j];
        }
        if (!GATHER)
            asm volatile("s_wait_asynccnt 0x0" ::: "memory");
        __syncthreads();

        // ---- fragments per ISA VGPR layouts ----
        // A (16-bit 16x32): lane<16 holds K {0..7,16..23}; lane>=16 {8..15,24..31}
        // B staged as sB[n][k]: lane<16 holds K {0..7,16..23}; lane>=16 {8..15,24..31}
        const int kofA = hiHalf ? 8 : 0;
        const int kofB = hiHalf ? 8 : 0;
        FragAB fa[2], fb[4];
        #pragma unroll
        for (int mi = 0; mi < 2; ++mi) {
            const __bf16* pa = &sA[(wm + mi * 16 + lmod) * LSTR + kofA];
            fa[mi].q[0] = *(const uint4*)pa;
            fa[mi].q[1] = *(const uint4*)(pa + 16);
        }
        #pragma unroll
        for (int ni = 0; ni < 4; ++ni) {
            const __bf16* pb = &sB[(wn + ni * 16 + lmod) * LSTR + kofB];
            fb[ni].q[0] = *(const uint4*)pb;
            fb[ni].q[1] = *(const uint4*)(pb + 16);
        }
        #pragma unroll
        for (int mi = 0; mi < 2; ++mi)
            #pragma unroll
            for (int ni = 0; ni < 4; ++ni)
                acc[mi][ni] = __builtin_amdgcn_wmma_f32_16x16x32_bf16(
                    false, fa[mi].v, false, fb[ni].v,
                    (short)0, acc[mi][ni], false, false);
    }

    // ---- epilogue (C/D layout: VGPR r -> row r + (lane>=16 ? 8:0), col = lane&15) ----
    const int rowOff = hiHalf ? 8 : 0;
    #pragma unroll
    for (int mi = 0; mi < 2; ++mi) {
        #pragma unroll
        for (int ni = 0; ni < 4; ++ni) {
            int gcol = tileN + wn + ni * 16 + lmod;
            float bv = (gcol < Nsize) ? bb[gcol] : 0.0f;
            #pragma unroll
            for (int r = 0; r < 8; ++r) {
                int grow = tileM + wm + mi * 16 + rowOff + r;
                float v = acc[mi][ni][r] + bv;
                if (MODE == 0) v = fmaxf(v, 0.0f);
                if (MODE == 2) v = (v > 0.0f) ? v : 0.2f * v;
                if (MODE == 3) {
                    if (gcol < Nsize) {
                        float ov = valid[grow] ? v : 0.0f;
                        ((float*)OutV)[(size_t)bz * strideO + (size_t)grow * ldo + gcol] = ov;
                    }
                } else {
                    if (gcol < Nsize)
                        ((__bf16*)OutV)[(size_t)bz * strideO + (size_t)grow * ldo + gcol] = (__bf16)v;
                }
            }
        }
    }
}

// =====================================================================
// Router: fp32 logits (routing decisions must match the reference),
// softmax, top-2 with smaller-index tie-break, normalized gates.
// One wave per token; router_w (24 KB) cached in LDS.
// =====================================================================
__global__ __launch_bounds__(256)
void router_kernel(const float* __restrict__ x, const int* __restrict__ task_in,
                   const float* __restrict__ temb, const float* __restrict__ rw,
                   float* __restrict__ probs, float* __restrict__ gate,
                   int* __restrict__ eidx)
{
    __shared__ float srw[DIM * NEXP];
    for (int i = threadIdx.x; i < DIM * NEXP; i += 256) srw[i] = rw[i];
    __syncthreads();

    const int wave = threadIdx.x >> 5;
    const int lane = threadIdx.x & 31;
    const int t = blockIdx.x * 8 + wave;

    float a[NEXP];
    #pragma unroll
    for (int e = 0; e < NEXP; ++e) a[e] = 0.0f;

    const int task = task_in[t];
    const float* xp = x + (size_t)t * DIM;
    const float* tp = temb + (size_t)task * DIM;
    for (int d = lane; d < DIM; d += 32) {
        float xv = xp[d] + tp[d];
        const float* w = &srw[d * NEXP];
        #pragma unroll
        for (int e = 0; e < NEXP; ++e) a[e] += xv * w[e];
    }
    #pragma unroll
    for (int e = 0; e < NEXP; ++e)
        for (int off = 16; off > 0; off >>= 1) a[e] += __shfl_xor(a[e], off, 32);

    if (lane == 0) {
        float m = a[0];
        #pragma unroll
        for (int e = 1; e < NEXP; ++e) m = fmaxf(m, a[e]);
        float p[NEXP], Z = 0.0f;
        #pragma unroll
        for (int e = 0; e < NEXP; ++e) { p[e] = expf(a[e] - m); Z += p[e]; }
        float inv = 1.0f / Z;
        #pragma unroll
        for (int e = 0; e < NEXP; ++e) { p[e] *= inv; probs[(size_t)t * NEXP + e] = p[e]; }
        int i1 = 0; float v1 = p[0];
        #pragma unroll
        for (int e = 1; e < NEXP; ++e) if (p[e] > v1) { v1 = p[e]; i1 = e; }
        int i2 = -1; float v2 = -1.0f;
        #pragma unroll
        for (int e = 0; e < NEXP; ++e) if (e != i1 && p[e] > v2) { v2 = p[e]; i2 = e; }
        float invs = 1.0f / (v1 + v2);
        gate[t * 2 + 0] = v1 * invs;  gate[t * 2 + 1] = v2 * invs;
        eidx[t * 2 + 0] = i1;         eidx[t * 2 + 1] = i2;
    }
}

// Deterministic (fixed-order) reduction for aux loss; also zeroes capacity counters.
__global__ __launch_bounds__(256)
void aux_reduce_kernel(const float* __restrict__ probs, const int* __restrict__ eidx,
                       float* __restrict__ aux_out, int* __restrict__ cap_cnt)
{
    __shared__ float sp[256 * NEXP];
    __shared__ int   sc[256 * NEXP];
    __shared__ float fin[NEXP];
    __shared__ int   finc[NEXP];
    const int tid = threadIdx.x;
    float ps[NEXP]; int cs[NEXP];
    #pragma unroll
    for (int e = 0; e < NEXP; ++e) { ps[e] = 0.0f; cs[e] = 0; }
    for (int t = tid; t < N_TOK; t += 256)
        #pragma unroll
        for (int e = 0; e < NEXP; ++e) ps[e] += probs[(size_t)t * NEXP + e];
    for (int i = tid; i < N_TOK * TOPK; i += 256) cs[eidx[i]]++;
    #pragma unroll
    for (int e = 0; e < NEXP; ++e) { sp[e * 256 + tid] = ps[e]; sc[e * 256 + tid] = cs[e]; }
    __syncthreads();
    if (tid < NEXP) {
        float s = 0.0f; int c = 0;
        for (int j = 0; j < 256; ++j) { s += sp[tid * 256 + j]; c += sc[tid * 256 + j]; }
        fin[tid] = s; finc[tid] = c;
    }
    __syncthreads();
    if (tid == 0) {
        float aux = 0.0f;
        for (int e = 0; e < NEXP; ++e)
            aux += ((float)finc[e] / (float)(N_TOK * TOPK)) * (fin[e] / (float)N_TOK);
        aux_out[0] = aux * (float)NEXP;
    }
    if (tid < NEXP) cap_cnt[tid] = 0;
}

// Atomic slot assignment (per-row expert math is slot-invariant; capacity
// never binds in expectation: 4096 << 5120).
__global__ void slot_assign_kernel(const int* __restrict__ eidx, int* __restrict__ cnt,
                                   int* __restrict__ slot)
{
    int i = blockIdx.x * blockDim.x + threadIdx.x;
    if (i >= N_TOK * TOPK) return;
    int e = eidx[i];
    int s = atomicAdd(&cnt[e], 1);
    slot[i] = (s < CAP) ? s : -1;
}

// Scatter x rows (fp32 -> bf16) into dispatch buffer.
__global__ __launch_bounds__(192)
void scatter_kernel(const float* __restrict__ x, const int* __restrict__ eidx,
                    const int* __restrict__ slot, __bf16* __restrict__ buf)
{
    const int i = blockIdx.x;
    const int s = slot[i];
    if (s < 0) return;
    const int e = eidx[i];
    const int t = i >> 1;
    const float* src = x + (size_t)t * DIM;
    __bf16* dst = buf + ((size_t)e * CAP + s) * DIM;
    const int d = threadIdx.x * 4;
    float4 v = *(const float4*)(src + d);
    __bf16 o[4] = {(__bf16)v.x, (__bf16)v.y, (__bf16)v.z, (__bf16)v.w};
    *(uint2*)(dst + d) = *(const uint2*)o;
}

// Gated combine of the two expert outputs per token.
__global__ __launch_bounds__(192)
void combine_kernel(const __bf16* __restrict__ eo, const int* __restrict__ eidx,
                    const int* __restrict__ slot, const float* __restrict__ gate,
                    float* __restrict__ y)
{
    const int t = blockIdx.x;
    const int d = threadIdx.x * 4;
    float4 r = {0.f, 0.f, 0.f, 0.f};
    #pragma unroll
    for (int k = 0; k < TOPK; ++k) {
        int s = slot[t * 2 + k];
        if (s < 0) continue;
        int e = eidx[t * 2 + k];
        float g = gate[t * 2 + k];
        const __bf16* p = eo + ((size_t)e * CAP + s) * DIM + d;
        __bf16 v[4]; *(uint2*)v = *(const uint2*)p;
        r.x += g * (float)v[0]; r.y += g * (float)v[1];
        r.z += g * (float)v[2]; r.w += g * (float)v[3];
    }
    *(float4*)(y + (size_t)t * DIM + d) = r;
}

__global__ void cvt_bf16_kernel(const float* __restrict__ src, __bf16* __restrict__ dst, int n)
{
    int i = blockIdx.x * blockDim.x + threadIdx.x;
    int stride = gridDim.x * blockDim.x;
    for (; i < n; i += stride) dst[i] = (__bf16)src[i];
}

// Stable single-block compaction per task (matches jnp.nonzero order).
// Writes gather indices, valid flags, and the float index outputs (incl. -1 pads).
__global__ __launch_bounds__(256)
void task_scan_kernel(const int* __restrict__ task_in, int* __restrict__ tidx,
                      int* __restrict__ tvalid, float* __restrict__ idx_out)
{
    __shared__ int scnt[3][256];
    __shared__ int total[3];
    const int tid = threadIdx.x;
    const int per = N_TOK / 256;
    int c[3] = {0, 0, 0};
    int base = tid * per;
    for (int j = 0; j < per; ++j) c[task_in[base + j]]++;
    for (int i = 0; i < 3; ++i) scnt[i][tid] = c[i];
    __syncthreads();
    if (tid == 0) {
        for (int i = 0; i < 3; ++i) {
            int run = 0;
            for (int t = 0; t < 256; ++t) { int v = scnt[i][t]; scnt[i][t] = run; run += v; }
            total[i] = run;
        }
    }
    __syncthreads();
    int pos[3] = {scnt[0][tid], scnt[1][tid], scnt[2][tid]};
    for (int j = 0; j < per; ++j) {
        int t = base + j;
        int tk = task_in[t];
        int p = pos[tk]++;
        tidx[tk * N_TOK + p] = t;
        idx_out[(size_t)tk * N_TOK + p] = (float)t;
    }
    __syncthreads();
    for (int i = 0; i < 3; ++i) {
        for (int p = total[i] + tid; p < N_TOK; p += 256) {
            tidx[i * N_TOK + p] = 0;
            idx_out[(size_t)i * N_TOK + p] = -1.0f;
        }
        for (int p = tid; p < N_TOK; p += 256)
            tvalid[i * N_TOK + p] = (p < total[i]) ? 1 : 0;
    }
}

// =====================================================================
extern "C" void kernel_launch(void* const* d_in, const int* in_sizes, int n_in,
                              void* d_out, int out_size, void* d_ws, size_t ws_size,
                              hipStream_t stream)
{
    (void)in_sizes; (void)n_in; (void)out_size; (void)ws_size;
    const float* x       = (const float*)d_in[0];
    const int*   task_in = (const int*)d_in[1];
    const float* temb    = (const float*)d_in[2];
    const float* rw      = (const float*)d_in[3];
    const float* ew1     = (const float*)d_in[4];
    const float* eb1     = (const float*)d_in[5];
    const float* ew2     = (const float*)d_in[6];
    const float* eb2     = (const float*)d_in[7];
    const float* tw0[3]  = {(const float*)d_in[8],  (const float*)d_in[12], (const float*)d_in[16]};
    const float* tb0[3]  = {(const float*)d_in[9],  (const float*)d_in[13], (const float*)d_in[17]};
    const float* tw1[3]  = {(const float*)d_in[10], (const float*)d_in[14], (const float*)d_in[18]};
    const float* tb1[3]  = {(const float*)d_in[11], (const float*)d_in[15], (const float*)d_in[19]};
    const int tdim[3] = {1, 101, 1000};

    // ---- workspace layout ----
    char* ws = (char*)d_ws;
    size_t off = 0;
    auto take = [&](size_t bytes) { size_t r = off; off += (bytes + 255) & ~(size_t)255; return r; };
    __bf16* buf   = (__bf16*)(ws + take((size_t)NEXP * CAP * DIM * 2));   // dispatch, later reused as eo
    __bf16* hbuf  = (__bf16*)(ws + take((size_t)NEXP * CAP * DIM * 2));
    __bf16* w1b   = (__bf16*)(ws + take((size_t)NEXP * DIM * DIM * 2));
    __bf16* w2b   = (__bf16*)(ws + take((size_t)NEXP * DIM * DIM * 2));
    __bf16* tw0b  = (__bf16*)(ws + take((size_t)3 * DIM * DHALF * 2));
    __bf16* tw1b  = (__bf16*)(ws + take((size_t)DHALF * (1 + 101 + 1000) * 2));
    float*  yb    = (float*) (ws + take((size_t)N_TOK * DIM * 4));
    float*  probs = (float*) (ws + take((size_t)N_TOK * NEXP * 4));
    float*  gate  = (float*) (ws + take((size_t)N_TOK * TOPK * 4));
    int*    eidx  = (int*)   (ws + take((size_t)N_TOK * TOPK * 4));
    int*    slot  = (int*)   (ws + take((size_t)N_TOK * TOPK * 4));
    int*    cnt   = (int*)   (ws + take((size_t)NEXP * 4));
    int*    tidx  = (int*)   (ws + take((size_t)3 * N_TOK * 4));
    int*    tval  = (int*)   (ws + take((size_t)3 * N_TOK * 4));
    __bf16* thid  = (__bf16*)(ws + take((size_t)N_TOK * DHALF * 2));

    float* out = (float*)d_out;
    const size_t AUX_POS = (size_t)N_TOK * 1105;              // 3N idx + N*1102 outs
    const size_t oOff[3] = {(size_t)3 * N_TOK, (size_t)4 * N_TOK, (size_t)105 * N_TOK};
    const size_t tw1off[3] = {0, (size_t)DHALF * 1, (size_t)DHALF * (1 + 101)};

    // ---- weight conversions (fp32 -> bf16) ----
    cvt_bf16_kernel<<<2048, 256, 0, stream>>>(ew1, w1b, NEXP * DIM * DIM);
    cvt_bf16_kernel<<<2048, 256, 0, stream>>>(ew2, w2b, NEXP * DIM * DIM);
    for (int i = 0; i < 3; ++i) {
        cvt_bf16_kernel<<<512, 256, 0, stream>>>(tw0[i], tw0b + (size_t)i * DIM * DHALF, DIM * DHALF);
        cvt_bf16_kernel<<<256, 256, 0, stream>>>(tw1[i], tw1b + tw1off[i], DHALF * tdim[i]);
    }

    // ---- routing (fp32) + aux loss + dispatch ----
    router_kernel<<<N_TOK / 8, 256, 0, stream>>>(x, task_in, temb, rw, probs, gate, eidx);
    aux_reduce_kernel<<<1, 256, 0, stream>>>(probs, eidx, out + AUX_POS, cnt);
    slot_assign_kernel<<<(N_TOK * TOPK + 255) / 256, 256, 0, stream>>>(eidx, cnt, slot);
    scatter_kernel<<<N_TOK * TOPK, 192, 0, stream>>>(x, eidx, slot, buf);

    // ---- expert FFN: two batched bf16 WMMA GEMMs ----
    dim3 egrid(DIM / 128, CAP / 128, NEXP);
    wmma_gemm_kernel<0, false, true><<<egrid, 256, 0, stream>>>(
        buf, nullptr, nullptr, w1b, eb1, (void*)hbuf, nullptr,
        DIM, DIM, DIM, DIM, DIM,
        (long long)CAP * DIM, (long long)DIM * DIM, (long long)DIM, (long long)CAP * DIM,
        cnt, CAP);
    wmma_gemm_kernel<1, false, true><<<egrid, 256, 0, stream>>>(
        hbuf, nullptr, nullptr, w2b, eb2, (void*)buf /*eo reuses buf*/, nullptr,
        DIM, DIM, DIM, DIM, DIM,
        (long long)CAP * DIM, (long long)DIM * DIM, (long long)DIM, (long long)CAP * DIM,
        cnt, CAP);

    combine_kernel<<<N_TOK, 192, 0, stream>>>(buf, eidx, slot, gate, yb);

    // ---- per-task stable compaction + heads ----
    task_scan_kernel<<<1, 256, 0, stream>>>(task_in, tidx, tval, out);

    for (int i = 0; i < 3; ++i) {
        dim3 g0(DHALF / 128, N_TOK / 128, 1);
        wmma_gemm_kernel<2, true, true><<<g0, 256, 0, stream>>>(
            nullptr, yb, tidx + (size_t)i * N_TOK,
            tw0b + (size_t)i * DIM * DHALF, tb0[i], (void*)thid, nullptr,
            DHALF, DIM, DIM, DHALF, DHALF, 0, 0, 0, 0, nullptr, 0);
        int ntiles = (tdim[i] + 127) / 128;
        dim3 g1(ntiles, N_TOK / 128, 1);
        if ((tdim[i] & 7) == 0) {
            wmma_gemm_kernel<3, false, true><<<g1, 256, 0, stream>>>(
                thid, nullptr, nullptr, tw1b + tw1off[i], tb1[i],
                (void*)(out + oOff[i]), tval + (size_t)i * N_TOK,
                tdim[i], DHALF, DHALF, tdim[i], tdim[i], 0, 0, 0, 0, nullptr, 0);
        } else {
            wmma_gemm_kernel<3, false, false><<<g1, 256, 0, stream>>>(
                thid, nullptr, nullptr, tw1b + tw1off[i], tb1[i],
                (void*)(out + oOff[i]), tval + (size_t)i * N_TOK,
                tdim[i], DHALF, DHALF, tdim[i], tdim[i], 0, 0, 0, 0, nullptr, 0);
        }
    }
}
